// SpatialTransformer_1769526526711
// MI455X (gfx1250) — matrix operand order repaired and checked by reading, not compiled, not run on hardware
//
#include <hip/hip_runtime.h>
#include <hip/hip_bf16.h>
#include <math.h>

// ---------------- problem constants ----------------
#define Mtok   4096          // 64*64 tokens
#define Cch    320
#define NHEADS 8
#define DHEAD  40
#define DHP    64            // dh padded for QK^T k-dim
#define DVP    48            // dh padded for PV n-dim
#define CTXL   77
#define CTXLP  96            // cross keys padded to mult of 32
#define CTXD   768
#define FFI    1280

typedef __attribute__((ext_vector_type(16))) _Float16 v16h;
typedef __attribute__((ext_vector_type(8)))  float    v8f;

// ---------------- WMMA fragment helpers ----------------
// A-matrix 16x32 f16 layout (ISA 7.12.2): lane<16 holds K {0..7,16..23}, lane>=16 holds {8..15,24..31}
__device__ __forceinline__ v16h ldA(const _Float16* row, int lane) {
    int kb = (lane & 16) ? 8 : 0;
    union { float4 f[2]; v16h v; } u;
    u.f[0] = *(const float4*)(row + kb);
    u.f[1] = *(const float4*)(row + kb + 16);
    return u.v;
}
// B-matrix 32x16 f16 layout: lane = column, lanes 0-15 hold K {0..7,16..23}, lanes 16-31 hold K {8..15,24..31}
__device__ __forceinline__ v16h ldB(const _Float16* row, int lane) {
    int kb = (lane & 16) ? 8 : 0;
    union { float4 f[2]; v16h v; } u;
    u.f[0] = *(const float4*)(row + kb);
    u.f[1] = *(const float4*)(row + kb + 16);
    return u.v;
}
__device__ __forceinline__ v8f wmma(v16h a, v16h b, v8f c) {
    return __builtin_amdgcn_wmma_f32_16x16x32_f16(false, a, false, b, (short)0, c, false, false);
}
__device__ __forceinline__ float halfmax(float v) {
    for (int off = 1; off < 16; off <<= 1) v = fmaxf(v, __shfl_xor(v, off, 32));
    return v;
}
__device__ __forceinline__ float halfsum(float v) {
    for (int off = 1; off < 16; off <<= 1) v += __shfl_xor(v, off, 32);
    return v;
}

// ---------------- generic WMMA GEMM, 16x64 strip per wave ----------------
// out[M,N] = A[M,K](f16) @ BT[N,K]^T(f16) + bias + resid ; optional f32/f16 stores,
// optional transposed store outT[n*M+m] += residT (final bohw projection).
// N must be a multiple of 64. A fragment is reused across the 4 N-tiles of the strip.
__global__ __launch_bounds__(128)
void gemm_wmma(const _Float16* __restrict__ A, const _Float16* __restrict__ BT,
               const float* __restrict__ bias, const float* __restrict__ resid,
               float* __restrict__ outF, _Float16* __restrict__ outH,
               float* __restrict__ outT, const float* __restrict__ residT,
               int M, int N, int K) {
    int lane = threadIdx.x & 31;
    int wave = blockIdx.x * (blockDim.x >> 5) + (threadIdx.x >> 5);
    int strips = N >> 6;
    int tilesM = (M + 15) >> 4;
    if (wave >= tilesM * strips) return;
    int mt = wave / strips, st = wave % strips;
    int rA = mt * 16 + (lane & 15); if (rA > M - 1) rA = M - 1;   // clamp for M % 16 != 0
    const _Float16* aRow = A + (size_t)rA * K;
    const _Float16* b0 = BT + (size_t)(st * 64 + (lane & 15)) * K;
    const _Float16* b1 = b0 + (size_t)16 * K;
    const _Float16* b2 = b0 + (size_t)32 * K;
    const _Float16* b3 = b0 + (size_t)48 * K;
    v8f acc[4] = {};
    for (int k0 = 0; k0 < K; k0 += 32) {
        v16h fa = ldA(aRow + k0, lane);      // shared across 4 N-tiles
        acc[0] = wmma(fa, ldB(b0 + k0, lane), acc[0]);
        acc[1] = wmma(fa, ldB(b1 + k0, lane), acc[1]);
        acc[2] = wmma(fa, ldB(b2 + k0, lane), acc[2]);
        acc[3] = wmma(fa, ldB(b3 + k0, lane), acc[3]);
    }
    int rbase = mt * 16 + ((lane & 16) ? 8 : 0);
#pragma unroll
    for (int t = 0; t < 4; t++) {
        int col = st * 64 + t * 16 + (lane & 15);
        float bv = bias ? bias[col] : 0.f;
#pragma unroll
        for (int r = 0; r < 8; r++) {
            int row = rbase + r;
            if (row >= M) continue;
            float v = acc[t][r] + bv;
            if (resid) v += resid[(size_t)row * N + col];
            if (outF)  outF[(size_t)row * N + col] = v;
            if (outH)  outH[(size_t)row * N + col] = (_Float16)v;
            if (outT) {
                size_t o = (size_t)col * M + row;
                outT[o] = v + (residT ? residT[o] : 0.f);
            }
        }
    }
}

// ---------------- fused GEGLU GEMM: g = a * gelu(gate) ----------------
// p = A @ BT^T + bias, with BT holding 2*FFI rows; col n -> a, col n+FFI -> gate.
__global__ __launch_bounds__(128)
void gemm_geglu(const _Float16* __restrict__ A, const _Float16* __restrict__ BT,
                const float* __restrict__ bias, _Float16* __restrict__ outH,
                int M, int Npair, int K) {
    int lane = threadIdx.x & 31;
    int wave = blockIdx.x * (blockDim.x >> 5) + (threadIdx.x >> 5);
    int tilesN = Npair >> 4;
    if (wave >= (M >> 4) * tilesN) return;
    int mt = wave / tilesN, nt = wave % tilesN;
    int rA = mt * 16 + (lane & 15);
    int rB = nt * 16 + (lane & 15);
    const _Float16* aRow = A + (size_t)rA * K;
    const _Float16* bRowA = BT + (size_t)rB * K;
    const _Float16* bRowG = BT + (size_t)(rB + Npair) * K;
    v8f accA = {}, accG = {};
    for (int k0 = 0; k0 < K; k0 += 32) {
        v16h fa = ldA(aRow + k0, lane);      // shared by a / gate tiles
        accA = wmma(fa, ldB(bRowA + k0, lane), accA);
        accG = wmma(fa, ldB(bRowG + k0, lane), accG);
    }
    int col   = nt * 16 + (lane & 15);
    int rbase = mt * 16 + ((lane & 16) ? 8 : 0);
    float ba = bias[col], bg = bias[col + Npair];
#pragma unroll
    for (int r = 0; r < 8; r++) {
        int row = rbase + r;
        float a = accA[r] + ba;
        float g = accG[r] + bg;
        float ge = 0.5f * g * (1.f + erff(g * 0.70710678118654752f));  // exact gelu
        outH[(size_t)row * Npair + col] = (_Float16)(a * ge);
    }
}

// ---------------- flash attention (WMMA, online softmax) ----------------
// Qh: [h][Lq][DHP]  Kh: [h][KR][DHP]  VT: [h][DVP][ldV]
// outH: [i][d*8+h] f16 (reference's (dh,heads) channel interleave), d < DHEAD
__global__ __launch_bounds__(128)
void flash_wmma(const _Float16* __restrict__ Qh, const _Float16* __restrict__ Kh,
                const _Float16* __restrict__ VT, _Float16* __restrict__ outH,
                int Lq, int Lk, int LkPad, int KR, int ldV) {
    __shared__ _Float16 pbuf[4][16 * 32];
    int lane  = threadIdx.x & 31;
    int wslot = threadIdx.x >> 5;
    int wave  = blockIdx.x * 4 + wslot;
    int qtiles = Lq >> 4;
    int h  = wave / qtiles;
    int mt = wave % qtiles;
    int hi = (lane & 16) ? 8 : 0;

    int rQ = mt * 16 + (lane & 15);
    const _Float16* qRow = Qh + ((size_t)h * Lq + rQ) * DHP;
    v16h qa0 = ldA(qRow, lane);
    v16h qa1 = ldA(qRow + 32, lane);

    float m[8], l[8];
#pragma unroll
    for (int r = 0; r < 8; r++) { m[r] = -1e30f; l[r] = 0.f; }
    v8f o0 = {}, o1 = {}, o2 = {};

    for (int j0 = 0; j0 < LkPad; j0 += 32) {
        // S = Q K^T for 32 keys (two 16-key tiles), K-dim = DHP
        v8f s0 = {}, s1 = {};
        {
            const _Float16* kRow = Kh + ((size_t)h * KR + j0 + (lane & 15)) * DHP;
            s0 = wmma(qa0, ldB(kRow, lane), s0);
            s0 = wmma(qa1, ldB(kRow + 32, lane), s0);
            kRow += 16 * DHP;
            s1 = wmma(qa0, ldB(kRow, lane), s1);
            s1 = wmma(qa1, ldB(kRow + 32, lane), s1);
        }
        bool v0 = (j0 + (lane & 15))      < Lk;
        bool v1 = (j0 + 16 + (lane & 15)) < Lk;
#pragma unroll
        for (int r = 0; r < 8; r++) {
            float a  = v0 ? s0[r] : -1e30f;
            float b  = v1 ? s1[r] : -1e30f;
            float mx = halfmax(fmaxf(a, b));            // row max over 32 keys
            float nm = fmaxf(m[r], mx);
            float sc = __expf(m[r] - nm);
            float p0 = v0 ? __expf(s0[r] - nm) : 0.f;
            float p1 = v1 ? __expf(s1[r] - nm) : 0.f;
            l[r] = l[r] * sc + halfsum(p0 + p1);
            m[r] = nm;
            o0[r] *= sc; o1[r] *= sc; o2[r] *= sc;
            pbuf[wslot][(r + hi) * 32 + (lane & 15)]      = (_Float16)p0;
            pbuf[wslot][(r + hi) * 32 + 16 + (lane & 15)] = (_Float16)p1;
        }
        __syncthreads();
        v16h pa = ldA(&pbuf[wslot][(lane & 15) * 32], lane);   // P in A-layout (16x32)
        __syncthreads();
        {
            const _Float16* vRow = VT + ((size_t)h * DVP + (lane & 15)) * ldV + j0;
            o0 = wmma(pa, ldB(vRow, lane), o0);
            o1 = wmma(pa, ldB(vRow + (size_t)16 * ldV, lane), o1);
            o2 = wmma(pa, ldB(vRow + (size_t)32 * ldV, lane), o2);
        }
    }
#pragma unroll
    for (int r = 0; r < 8; r++) {
        float inv = 1.f / l[r];
        int row = mt * 16 + r + hi;
        int n = lane & 15;
        outH[(size_t)row * Cch + n * 8 + h]        = (_Float16)(o0[r] * inv);      // d = n
        outH[(size_t)row * Cch + (16 + n) * 8 + h] = (_Float16)(o1[r] * inv);      // d = 16+n
        if (32 + n < DHEAD)
            outH[(size_t)row * Cch + (32 + n) * 8 + h] = (_Float16)(o2[r] * inv);  // d = 32+n
    }
}

// ---------------- GroupNorm (32 groups over 320ch x 4096) -> token-major f16 ----------------
__global__ __launch_bounds__(256)
void groupnorm_k(const float* __restrict__ x, const float* __restrict__ w,
                 const float* __restrict__ b, _Float16* __restrict__ outA) {
    const int CPG = Cch / 32, HW = Mtok, NE = CPG * HW;
    __shared__ float s1[256], s2[256];
    int g = blockIdx.x, tid = threadIdx.x;
    float sum = 0.f, sq = 0.f;
    for (int e = tid; e < NE; e += 256) {
        float v = x[(size_t)(g * CPG + e / HW) * HW + (e % HW)];
        sum += v; sq += v * v;
    }
    s1[tid] = sum; s2[tid] = sq; __syncthreads();
    for (int s = 128; s > 0; s >>= 1) {
        if (tid < s) { s1[tid] += s1[tid + s]; s2[tid] += s2[tid + s]; }
        __syncthreads();
    }
    float mean = s1[0] / NE;
    float rstd = rsqrtf(s2[0] / NE - mean * mean + 1e-6f);
    for (int e = tid; e < NE; e += 256) {
        int c = g * CPG + e / HW, i = e % HW;
        float v = (x[(size_t)c * HW + i] - mean) * rstd * w[c] + b[c];
        outA[(size_t)i * Cch + c] = (_Float16)v;
    }
}

// ---------------- LayerNorm over C=320, one block per token ----------------
__global__ __launch_bounds__(128)
void layernorm_k(const float* __restrict__ t, const float* __restrict__ w,
                 const float* __restrict__ b, _Float16* __restrict__ out) {
    __shared__ float s1[128], s2[128];
    int row = blockIdx.x, tid = threadIdx.x;
    const float* p = t + (size_t)row * Cch;
    float sum = 0.f, sq = 0.f;
    for (int c = tid; c < Cch; c += 128) { float v = p[c]; sum += v; sq += v * v; }
    s1[tid] = sum; s2[tid] = sq; __syncthreads();
    for (int s = 64; s > 0; s >>= 1) {
        if (tid < s) { s1[tid] += s1[tid + s]; s2[tid] += s2[tid + s]; }
        __syncthreads();
    }
    float mean = s1[0] / Cch;
    float rstd = rsqrtf(s2[0] / Cch - mean * mean + 1e-5f);
    for (int c = tid; c < Cch; c += 128)
        out[(size_t)row * Cch + c] = (_Float16)((p[c] - mean) * rstd * w[c] + b[c]);
}

// ---------------- weight transpose f32(K,N) -> f16(N,K) ----------------
__global__ void transpose_f16(const float* __restrict__ W, _Float16* __restrict__ WT, int K, int N) {
    int idx = blockIdx.x * 256 + threadIdx.x;
    if (idx >= K * N) return;
    int k = idx % K, n = idx / K;
    WT[(size_t)n * K + k] = (_Float16)W[(size_t)k * N + n];
}
__global__ void cvt_f16(const float* __restrict__ src, _Float16* __restrict__ dst, int n) {
    int idx = blockIdx.x * 256 + threadIdx.x;
    if (idx < n) dst[idx] = (_Float16)src[idx];
}

// ---------------- head packing (channel = d*8 + h interleave) ----------------
// dst[h][i<Lpad][DHP] from src[i<Lvalid][Cch], zero-padded, optional scale (for Q)
__global__ void pack_rows(const float* __restrict__ src, _Float16* __restrict__ dst,
                          float scale, int Lpad, int Lvalid) {
    int idx = blockIdx.x * 256 + threadIdx.x;
    int total = NHEADS * Lpad * DHP;
    if (idx >= total) return;
    int d = idx % DHP, i = (idx / DHP) % Lpad, h = idx / (DHP * Lpad);
    float v = (d < DHEAD && i < Lvalid) ? src[(size_t)i * Cch + d * 8 + h] * scale : 0.f;
    dst[idx] = (_Float16)v;
}
// dst[h][d<DVP][j<Lpad] = V[j][d*8+h], zero-padded
__global__ void pack_vT(const float* __restrict__ src, _Float16* __restrict__ dst,
                        int Lpad, int Lvalid) {
    int idx = blockIdx.x * 256 + threadIdx.x;
    int total = NHEADS * DVP * Lpad;
    if (idx >= total) return;
    int j = idx % Lpad, d = (idx / Lpad) % DVP, h = idx / (Lpad * DVP);
    float v = (d < DHEAD && j < Lvalid) ? src[(size_t)j * Cch + d * 8 + h] : 0.f;
    dst[idx] = (_Float16)v;
}

// ---------------- host orchestration ----------------
static inline void launch_gemm(const _Float16* A, const _Float16* BT, const float* bias,
                               const float* resid, float* outF, _Float16* outH,
                               float* outT, const float* residT,
                               int M, int N, int K, hipStream_t s) {
    int tiles = ((M + 15) / 16) * (N / 64);     // 16x64 strip per wave
    gemm_wmma<<<(tiles + 3) / 4, 128, 0, s>>>(A, BT, bias, resid, outF, outH, outT, residT, M, N, K);
}

extern "C" void kernel_launch(void* const* d_in, const int* in_sizes, int n_in,
                              void* d_out, int out_size, void* d_ws, size_t ws_size,
                              hipStream_t stream) {
    (void)in_sizes; (void)n_in; (void)out_size; (void)ws_size;
    const float* x      = (const float*)d_in[0];
    const float* ctx    = (const float*)d_in[1];
    const float* gn_w   = (const float*)d_in[2];
    const float* gn_b   = (const float*)d_in[3];
    const float* w_in   = (const float*)d_in[4];
    const float* b_in   = (const float*)d_in[5];
    const float* ln1_w  = (const float*)d_in[6];
    const float* ln1_b  = (const float*)d_in[7];
    const float* wq1    = (const float*)d_in[8];
    const float* wk1    = (const float*)d_in[9];
    const float* wv1    = (const float*)d_in[10];
    const float* wo1    = (const float*)d_in[11];
    const float* bo1    = (const float*)d_in[12];
    const float* ln2_w  = (const float*)d_in[13];
    const float* ln2_b  = (const float*)d_in[14];
    const float* wq2    = (const float*)d_in[15];
    const float* wk2    = (const float*)d_in[16];
    const float* wv2    = (const float*)d_in[17];
    const float* wo2    = (const float*)d_in[18];
    const float* bo2    = (const float*)d_in[19];
    const float* ln3_w  = (const float*)d_in[20];
    const float* ln3_b  = (const float*)d_in[21];
    const float* wff1   = (const float*)d_in[22];
    const float* bff1   = (const float*)d_in[23];
    const float* wff2   = (const float*)d_in[24];
    const float* bff2   = (const float*)d_in[25];
    const float* w_out  = (const float*)d_in[26];
    const float* b_out  = (const float*)d_in[27];
    float* out = (float*)d_out;

    // workspace carve-up
    char* wsp = (char*)d_ws;
    auto alloc = [&](size_t bytes) -> void* {
        void* p = wsp; wsp += (bytes + 255) & ~(size_t)255; return p;
    };
    float*     t_f32   = (float*)alloc((size_t)Mtok * Cch * 4);
    float*     q_f32   = (float*)alloc((size_t)Mtok * Cch * 4);
    float*     k_f32   = (float*)alloc((size_t)Mtok * Cch * 4);
    float*     v_f32   = (float*)alloc((size_t)Mtok * Cch * 4);
    _Float16*  hA_f16  = (_Float16*)alloc((size_t)Mtok * Cch * 2);
    _Float16*  hn_f16  = (_Float16*)alloc((size_t)Mtok * Cch * 2);
    _Float16*  q_f16   = (_Float16*)alloc((size_t)Mtok * Cch * 2);
    _Float16*  attn_f16= (_Float16*)alloc((size_t)Mtok * Cch * 2);
    _Float16*  t_f16   = (_Float16*)alloc((size_t)Mtok * Cch * 2);
    _Float16*  g_f16   = (_Float16*)alloc((size_t)Mtok * FFI * 2);
    _Float16*  qh      = (_Float16*)alloc((size_t)NHEADS * Mtok * DHP * 2);
    _Float16*  kh      = (_Float16*)alloc((size_t)NHEADS * Mtok * DHP * 2);
    _Float16*  vT      = (_Float16*)alloc((size_t)NHEADS * DVP * Mtok * 2);
    _Float16*  ctx16   = (_Float16*)alloc((size_t)CTXL * CTXD * 2);
    float*     kc_f32  = (float*)alloc((size_t)CTXL * Cch * 4);
    float*     vc_f32  = (float*)alloc((size_t)CTXL * Cch * 4);
    _Float16*  kch     = (_Float16*)alloc((size_t)NHEADS * CTXLP * DHP * 2);
    _Float16*  vTc     = (_Float16*)alloc((size_t)NHEADS * DVP * CTXLP * 2);
    _Float16*  w_inT   = (_Float16*)alloc((size_t)Cch * Cch * 2);
    _Float16*  wq1T    = (_Float16*)alloc((size_t)Cch * Cch * 2);
    _Float16*  wk1T    = (_Float16*)alloc((size_t)Cch * Cch * 2);
    _Float16*  wv1T    = (_Float16*)alloc((size_t)Cch * Cch * 2);
    _Float16*  wo1T    = (_Float16*)alloc((size_t)Cch * Cch * 2);
    _Float16*  wq2T    = (_Float16*)alloc((size_t)Cch * Cch * 2);
    _Float16*  wk2T    = (_Float16*)alloc((size_t)CTXD * Cch * 2);
    _Float16*  wv2T    = (_Float16*)alloc((size_t)CTXD * Cch * 2);
    _Float16*  wo2T    = (_Float16*)alloc((size_t)Cch * Cch * 2);
    _Float16*  wff1T   = (_Float16*)alloc((size_t)Cch * 2 * FFI * 2);
    _Float16*  wff2T   = (_Float16*)alloc((size_t)FFI * Cch * 2);
    _Float16*  w_outT  = (_Float16*)alloc((size_t)Cch * Cch * 2);

    auto xpose = [&](const float* W, _Float16* WT, int K, int N) {
        transpose_f16<<<(K * N + 255) / 256, 256, 0, stream>>>(W, WT, K, N);
    };
    // weights -> f16 transposed (N x K)
    xpose(w_in,  w_inT,  Cch, Cch);
    xpose(wq1,   wq1T,   Cch, Cch);
    xpose(wk1,   wk1T,   Cch, Cch);
    xpose(wv1,   wv1T,   Cch, Cch);
    xpose(wo1,   wo1T,   Cch, Cch);
    xpose(wq2,   wq2T,   Cch, Cch);
    xpose(wk2,   wk2T,   CTXD, Cch);
    xpose(wv2,   wv2T,   CTXD, Cch);
    xpose(wo2,   wo2T,   Cch, Cch);
    xpose(wff1,  wff1T,  Cch, 2 * FFI);
    xpose(wff2,  wff2T,  FFI, Cch);
    xpose(w_out, w_outT, Cch, Cch);
    cvt_f16<<<(CTXL * CTXD + 255) / 256, 256, 0, stream>>>(ctx, ctx16, CTXL * CTXD);

    const float scale = 0.158113883008419f;  // 40^-0.5

    // GroupNorm -> token-major f16
    groupnorm_k<<<32, 256, 0, stream>>>(x, gn_w, gn_b, hA_f16);
    // proj_in: t = hA @ w_in + b_in
    launch_gemm(hA_f16, w_inT, b_in, nullptr, t_f32, nullptr, nullptr, nullptr, Mtok, Cch, Cch, stream);

    // ---- self attention ----
    layernorm_k<<<Mtok, 128, 0, stream>>>(t_f32, ln1_w, ln1_b, hn_f16);
    launch_gemm(hn_f16, wq1T, nullptr, nullptr, q_f32, q_f16, nullptr, nullptr, Mtok, Cch, Cch, stream);
    launch_gemm(q_f16, wk1T, nullptr, nullptr, k_f32, nullptr, nullptr, nullptr, Mtok, Cch, Cch, stream);  // k = q @ wk1
    launch_gemm(q_f16, wv1T, nullptr, nullptr, v_f32, nullptr, nullptr, nullptr, Mtok, Cch, Cch, stream);  // v = q @ wv1
    pack_rows<<<(NHEADS * Mtok * DHP + 255) / 256, 256, 0, stream>>>(q_f32, qh, scale, Mtok, Mtok);
    pack_rows<<<(NHEADS * Mtok * DHP + 255) / 256, 256, 0, stream>>>(k_f32, kh, 1.f, Mtok, Mtok);
    pack_vT  <<<(NHEADS * DVP * Mtok + 255) / 256, 256, 0, stream>>>(v_f32, vT, Mtok, Mtok);
    flash_wmma<<<NHEADS * (Mtok / 16) / 4, 128, 0, stream>>>(qh, kh, vT, attn_f16,
                                                             Mtok, Mtok, Mtok, Mtok, Mtok);
    // t += attn @ wo1 + bo1
    launch_gemm(attn_f16, wo1T, bo1, t_f32, t_f32, nullptr, nullptr, nullptr, Mtok, Cch, Cch, stream);

    // ---- cross attention ----
    layernorm_k<<<Mtok, 128, 0, stream>>>(t_f32, ln2_w, ln2_b, hn_f16);
    launch_gemm(hn_f16, wq2T, nullptr, nullptr, q_f32, nullptr, nullptr, nullptr, Mtok, Cch, Cch, stream);
    launch_gemm(ctx16, wk2T, nullptr, nullptr, kc_f32, nullptr, nullptr, nullptr, CTXL, Cch, CTXD, stream);
    launch_gemm(ctx16, wv2T, nullptr, nullptr, vc_f32, nullptr, nullptr, nullptr, CTXL, Cch, CTXD, stream);
    pack_rows<<<(NHEADS * Mtok * DHP + 255) / 256, 256, 0, stream>>>(q_f32, qh, scale, Mtok, Mtok);
    pack_rows<<<(NHEADS * CTXLP * DHP + 255) / 256, 256, 0, stream>>>(kc_f32, kch, 1.f, CTXLP, CTXL);
    pack_vT  <<<(NHEADS * DVP * CTXLP + 255) / 256, 256, 0, stream>>>(vc_f32, vTc, CTXLP, CTXL);
    flash_wmma<<<NHEADS * (Mtok / 16) / 4, 128, 0, stream>>>(qh, kch, vTc, attn_f16,
                                                             Mtok, CTXL, CTXLP, CTXLP, CTXLP);
    launch_gemm(attn_f16, wo2T, bo2, t_f32, t_f32, nullptr, nullptr, nullptr, Mtok, Cch, Cch, stream);

    // ---- GEGLU feed-forward ----
    layernorm_k<<<Mtok, 128, 0, stream>>>(t_f32, ln3_w, ln3_b, hn_f16);
    {
        int tiles = (Mtok / 16) * (FFI / 16);
        gemm_geglu<<<(tiles + 3) / 4, 128, 0, stream>>>(hn_f16, wff1T, bff1, g_f16, Mtok, FFI, Cch);
    }
    // t += g @ wff2 + bff2 (also keep f16 copy of t for the final projection)
    launch_gemm(g_f16, wff2T, bff2, t_f32, t_f32, t_f16, nullptr, nullptr, Mtok, Cch, FFI, stream);

    // ---- proj_out + input residual, transposed store to (C, H*W) ----
    launch_gemm(t_f16, w_outT, b_out, nullptr, nullptr, nullptr, out, x, Mtok, Cch, Cch, stream);
}
